// GraphMSG_72593537237298
// MI455X (gfx1250) — matrix operand order repaired and checked by reading, not compiled, not run on hardware
//
#include <hip/hip_runtime.h>
#include <cstddef>
#include <cstdint>

typedef __bf16 bf16_t;
typedef __attribute__((ext_vector_type(16))) __bf16 v16bf;
typedef __attribute__((ext_vector_type(8)))  float  v8f;

struct Seg { const float* p; const int* gidx; int ld; int w; };

// ---------------------------------------------------------------------------
// GEMM: out[M x Nreal] = act( concat-A(M x Kpad) @ W(Kpad x N) + bias )
// A: virtual concat of up to NSEG segments (widths multiple of 32), optionally
//    row-gathered (fused graph gather). All A sources are padded to 128-row
//    multiples in workspace, so A loads need no guards (uniform EXEC).
// Wt: bf16 [Npad][Kpad] (transposed, zero-padded) -> 32B per-lane B loads.
// Each wave computes 16 x (NT*16) C; block = 8 waves = 128 x (NT*16).
// STAGEK>0: stage the NT*16-col weight panel in LDS once per block.
// K loop = one two-phase pipelined sub-loop per segment (no per-iteration
// segment selection; A address math is a single pointer increment).
// ---------------------------------------------------------------------------
template<int NSEG, int ACT, int STAGEK, int NT>
__global__ __launch_bounds__(256) void k_gemm(
    Seg s0, Seg s1, Seg s2,
    const unsigned short* __restrict__ Wt,
    const float* __restrict__ bias,
    float* __restrict__ out,
    int M, int Nreal, int Kpad, int ldOut)
{
  const int tid  = threadIdx.x;
  const int lane = tid & 31;
  const int wave = tid >> 5;
  const int row0 = (blockIdx.x * 8 + wave) * 16;
  const int col0 = blockIdx.y * (NT * 16);
  const int g = lane >> 4;
  const int l = lane & 15;

  __shared__ __align__(32) unsigned short lb[(STAGEK > 0) ? NT * 16 * STAGEK : 16];

  if (STAGEK > 0) {
    // NT*16-col weight panel is contiguous in Wt ([Npad][Kpad], Kpad==STAGEK)
    const uint4* src = (const uint4*)(Wt + (size_t)col0 * STAGEK);
    uint4* dst = (uint4*)lb;
    const int chunks = NT * 16 * STAGEK / 8;     // 16-byte chunks
    for (int i = tid; i < chunks; i += 256) dst[i] = src[i];
    __syncthreads();
  }

  const int ar = row0 + l;                 // A row fed by this lane (padded-safe)
  const int r0 = s0.gidx ? s0.gidx[ar] : ar;
  const int r1 = (NSEG >= 2) ? (s1.gidx ? s1.gidx[ar] : ar) : 0;
  const int r2 = (NSEG >= 3) ? (s2.gidx ? s2.gidx[ar] : ar) : 0;

  // per-lane hoisted A base pointers (include the 16g half-chunk offset)
  const float* ap0 = s0.p + (size_t)r0 * s0.ld + 16 * g;
  const float* ap1 = (NSEG >= 2) ? s1.p + (size_t)r1 * s1.ld + 16 * g : nullptr;
  const float* ap2 = (NSEG >= 3) ? s2.p + (size_t)r2 * s2.ld + 16 * g : nullptr;

  // per-lane B base + per-col-tile stride (constant -> folds into offsets)
  const unsigned short* wl =
      (STAGEK > 0) ? (const unsigned short*)lb + (size_t)l * STAGEK + 16 * g
                   : Wt + (size_t)(col0 + l) * Kpad + 16 * g;
  const size_t bts = (size_t)16 * ((STAGEK > 0) ? STAGEK : Kpad);

  auto loadA = [&](const float* ap, int kl, v16bf& af) {
    // A fragment: lane(g,l): row l,
    // K = [kl+16g, kl+16g+16) contiguous  (16g folded into ap)
    float4 c0 = *(const float4*)(ap + kl + 0);
    float4 c1 = *(const float4*)(ap + kl + 4);
    float4 c2 = *(const float4*)(ap + kl + 8);
    float4 c3 = *(const float4*)(ap + kl + 12);
    v16bf a;
    a[0]  = (bf16_t)c0.x; a[1]  = (bf16_t)c0.y; a[2]  = (bf16_t)c0.z; a[3]  = (bf16_t)c0.w;
    a[4]  = (bf16_t)c1.x; a[5]  = (bf16_t)c1.y; a[6]  = (bf16_t)c1.z; a[7]  = (bf16_t)c1.w;
    a[8]  = (bf16_t)c2.x; a[9]  = (bf16_t)c2.y; a[10] = (bf16_t)c2.z; a[11] = (bf16_t)c2.w;
    a[12] = (bf16_t)c3.x; a[13] = (bf16_t)c3.y; a[14] = (bf16_t)c3.z; a[15] = (bf16_t)c3.w;
    af = a;
  };
  auto loadB = [&](int kb, v16bf (&bf)[NT]) {
    // B fragment: lane(g,n=l): col, K = [kb+16g, kb+16g+16) contiguous bf16
#pragma unroll
    for (int t = 0; t < NT; ++t)
      bf[t] = *(const v16bf*)(wl + (size_t)t * bts + kb);
  };

  v8f acc[NT] = {};

  // one pipelined sub-loop per A segment (kbase = segment's global K offset)
  auto runSeg = [&](const float* ap, int width, int kbase) {
    const int nkb = width / 32;
    v16bf afA, bfA[NT];
    v16bf afB, bfB[NT];
    loadA(ap, 0, afA); loadB(kbase, bfA);
    int i = 0;
    for (; i + 2 <= nkb; i += 2) {
      loadA(ap, (i + 1) * 32, afB); loadB(kbase + (i + 1) * 32, bfB);
#pragma unroll
      for (int t = 0; t < NT; ++t)
        acc[t] = __builtin_amdgcn_wmma_f32_16x16x32_bf16(
                     false, afA, false, bfA[t], (short)0, acc[t], false, false);
      if (i + 2 < nkb) { loadA(ap, (i + 2) * 32, afA); loadB(kbase + (i + 2) * 32, bfA); }
#pragma unroll
      for (int t = 0; t < NT; ++t)
        acc[t] = __builtin_amdgcn_wmma_f32_16x16x32_bf16(
                     false, afB, false, bfB[t], (short)0, acc[t], false, false);
    }
    if (i < nkb) {   // odd tail (block already resident in A-phase buffers)
#pragma unroll
      for (int t = 0; t < NT; ++t)
        acc[t] = __builtin_amdgcn_wmma_f32_16x16x32_bf16(
                     false, afA, false, bfA[t], (short)0, acc[t], false, false);
    }
  };

  runSeg(ap0, s0.w, 0);
  if (NSEG >= 2) runSeg(ap1, s1.w, s0.w);
  if (NSEG >= 3) runSeg(ap2, s2.w, s0.w + s1.w);

  // C layout: acc[t][v] = C[row0 + v + 8g][col0 + 16t + l]
#pragma unroll
  for (int t = 0; t < NT; ++t) {
    const int colt = col0 + 16 * t + l;
    if (colt < Nreal) {
      const float b = bias[colt];
#pragma unroll
      for (int v = 0; v < 8; ++v) {
        const int row = row0 + v + 8 * g;
        if (row < M) {
          float y = acc[t][v] + b;
          if (ACT == 1) y = y * __builtin_amdgcn_rcpf(1.f + __expf(-y));  // SiLU
          out[(size_t)row * ldOut + colt] = y;
        }
      }
    }
  }
}

// W[K][N] f32 -> Wt[Npad][Kpad] bf16 (transpose + zero pad)
__global__ void k_convert_w(const float* __restrict__ W, unsigned short* __restrict__ Wt,
                            int K, int N, int Kpad, int Npad)
{
  int i = blockIdx.x * blockDim.x + threadIdx.x;
  int total = Kpad * Npad;
  if (i >= total) return;
  int n = i / Kpad, k = i % Kpad;
  float v = (k < K && n < N) ? W[(size_t)k * N + n] : 0.f;
  bf16_t b = (bf16_t)v;
  Wt[i] = __builtin_bit_cast(unsigned short, b);
}

// idx pad: dst[i] = i<E ? src[i] : 0
__global__ void k_pad_idx(const int* __restrict__ src, int* __restrict__ dst, int E, int Ep)
{
  int i = blockIdx.x * blockDim.x + threadIdx.x;
  if (i >= Ep) return;
  dst[i] = (i < E) ? src[i] : 0;
}

// row LayerNorm over 256 cols, one wave per row; optional residual add
__global__ __launch_bounds__(256) void k_layernorm(
    const float* __restrict__ x, const float* __restrict__ gamma,
    const float* __restrict__ beta, const float* __restrict__ res,
    float* __restrict__ out, int M)
{
  const int lane = threadIdx.x & 31;
  const int wave = threadIdx.x >> 5;
  const int row = blockIdx.x * 8 + wave;
  if (row >= M) return;
  const float* xr = x + (size_t)row * 256 + lane * 8;
  float4 a0 = *(const float4*)(xr + 0);
  float4 a1 = *(const float4*)(xr + 4);
  float s = a0.x + a0.y + a0.z + a0.w + a1.x + a1.y + a1.z + a1.w;
  for (int m = 16; m >= 1; m >>= 1) s += __shfl_xor(s, m, 32);
  const float mu = s * (1.f / 256.f);
  float q = 0.f;
  {
    float d;
    d = a0.x - mu; q += d * d;  d = a0.y - mu; q += d * d;
    d = a0.z - mu; q += d * d;  d = a0.w - mu; q += d * d;
    d = a1.x - mu; q += d * d;  d = a1.y - mu; q += d * d;
    d = a1.z - mu; q += d * d;  d = a1.w - mu; q += d * d;
  }
  for (int m = 16; m >= 1; m >>= 1) q += __shfl_xor(q, m, 32);
  const float rinv = __frsqrt_rn(q * (1.f / 256.f) + 1e-5f);
  const int c = lane * 8;
  float4 g0 = *(const float4*)(gamma + c);
  float4 g1 = *(const float4*)(gamma + c + 4);
  float4 b0 = *(const float4*)(beta + c);
  float4 b1 = *(const float4*)(beta + c + 4);
  float4 y0, y1;
  y0.x = (a0.x - mu) * rinv * g0.x + b0.x;  y0.y = (a0.y - mu) * rinv * g0.y + b0.y;
  y0.z = (a0.z - mu) * rinv * g0.z + b0.z;  y0.w = (a0.w - mu) * rinv * g0.w + b0.w;
  y1.x = (a1.x - mu) * rinv * g1.x + b1.x;  y1.y = (a1.y - mu) * rinv * g1.y + b1.y;
  y1.z = (a1.z - mu) * rinv * g1.z + b1.z;  y1.w = (a1.w - mu) * rinv * g1.w + b1.w;
  if (res) {
    const float* rr = res + (size_t)row * 256 + c;
    float4 r0 = *(const float4*)(rr + 0);
    float4 r1 = *(const float4*)(rr + 4);
    y0.x += r0.x; y0.y += r0.y; y0.z += r0.z; y0.w += r0.w;
    y1.x += r1.x; y1.y += r1.y; y1.z += r1.z; y1.w += r1.w;
  }
  float* orow = out + (size_t)row * 256 + c;
  *(float4*)(orow + 0) = y0;
  *(float4*)(orow + 4) = y1;
}

// e[row] += m[row];  agg[dst[row]] += m[row]  (segment-sum via f32 atomics)
__global__ __launch_bounds__(256) void k_edge_scatter(
    const float* __restrict__ m, float* __restrict__ e, float* __restrict__ agg,
    const int* __restrict__ dstIdx, int E)
{
  const int lane = threadIdx.x & 31;
  const int wave = threadIdx.x >> 5;
  const int edge = blockIdx.x * 8 + wave;
  if (edge >= E) return;
  const int d = dstIdx[edge];
  const int c = lane * 8;
  const float* mr = m + (size_t)edge * 256 + c;
  float* er = e + (size_t)edge * 256 + c;
  float* ar = agg + (size_t)d * 256 + c;
  float4 m0 = *(const float4*)(mr + 0);
  float4 m1 = *(const float4*)(mr + 4);
  float4 e0 = *(const float4*)(er + 0);
  float4 e1 = *(const float4*)(er + 4);
  e0.x += m0.x; e0.y += m0.y; e0.z += m0.z; e0.w += m0.w;
  e1.x += m1.x; e1.y += m1.y; e1.z += m1.z; e1.w += m1.w;
  *(float4*)(er + 0) = e0;
  *(float4*)(er + 4) = e1;
  atomicAdd(ar + 0, m0.x); atomicAdd(ar + 1, m0.y);
  atomicAdd(ar + 2, m0.z); atomicAdd(ar + 3, m0.w);
  atomicAdd(ar + 4, m1.x); atomicAdd(ar + 5, m1.y);
  atomicAdd(ar + 6, m1.z); atomicAdd(ar + 7, m1.w);
}

__global__ void k_zero(float* __restrict__ p, size_t n)
{
  size_t i = (size_t)blockIdx.x * blockDim.x + threadIdx.x;
  const size_t stride = (size_t)gridDim.x * blockDim.x;
  for (; i < n; i += stride) p[i] = 0.f;
}

// x_era: [NERA][224] = [ x(2 steps x 100) | sin/cos(coords) | era_tr(8) | 0-pad ]
__global__ void k_build_xera(const float* __restrict__ x, const float* __restrict__ coords,
                             const float* __restrict__ tr, float* __restrict__ out, int NERA)
{
  int i = blockIdx.x * blockDim.x + threadIdx.x;
  int total = NERA * 224;
  if (i >= total) return;
  int n = i / 224, c = i % 224;
  float v = 0.f;
  if (c < 200)      { int mi = c / 100, fi = c % 100; v = x[((size_t)mi * NERA + n) * 100 + fi]; }
  else if (c < 204) { int j = c - 200; float cc = coords[n * 2 + (j & 1)]; v = (j < 2) ? __sinf(cc) : __cosf(cc); }
  else if (c < 212) { v = tr[n * 8 + (c - 204)]; }
  out[i] = v;
}

// x_h: [NH][32] = [ sin/cos(coords_h) | h_tr(8) | 0-pad ]
__global__ void k_build_xh(const float* __restrict__ coords, const float* __restrict__ tr,
                           float* __restrict__ out, int NH)
{
  int i = blockIdx.x * blockDim.x + threadIdx.x;
  int total = NH * 32;
  if (i >= total) return;
  int n = i / 32, c = i % 32;
  float v = 0.f;
  if (c < 4)       { float cc = coords[n * 2 + (c & 1)]; v = (c < 2) ? __sinf(cc) : __cosf(cc); }
  else if (c < 12) { v = tr[n * 8 + (c - 4)]; }
  out[i] = v;
}

// edge_in: [E][32] = [ attr(3) | tr(8) | 0-pad ]
__global__ void k_build_edgein(const float* __restrict__ attr, const float* __restrict__ tr,
                               float* __restrict__ out, int E)
{
  int i = blockIdx.x * blockDim.x + threadIdx.x;
  int total = E * 32;
  if (i >= total) return;
  int e = i / 32, c = i % 32;
  float v = 0.f;
  if (c < 3)       v = attr[e * 3 + c];
  else if (c < 11) v = tr[e * 8 + (c - 3)];
  out[i] = v;
}

// noise_pad: [NH][32] = [ noise(4) | 0-pad ]
__global__ void k_pad_noise(const float* __restrict__ noise, float* __restrict__ out, int NH)
{
  int i = blockIdx.x * blockDim.x + threadIdx.x;
  int total = NH * 32;
  if (i >= total) return;
  int n = i / 32, c = i % 32;
  out[i] = (c < 4) ? noise[n * 4 + c] : 0.f;
}

// final: [NERA][96] -> d_out [NERA][88]
__global__ void k_copy_out(const float* __restrict__ src, float* __restrict__ dst, int NERA)
{
  int i = blockIdx.x * blockDim.x + threadIdx.x;
  int total = NERA * 88;
  if (i >= total) return;
  int n = i / 88, c = i % 88;
  dst[i] = src[(size_t)n * 96 + c];
}

// ---------------------------------------------------------------------------
extern "C" void kernel_launch(void* const* d_in, const int* in_sizes, int n_in,
                              void* d_out, int out_size, void* d_ws, size_t ws_size,
                              hipStream_t stream)
{
  (void)in_sizes; (void)n_in; (void)out_size; (void)ws_size;
  const int NERA = 40320, NH = 10242, EE2H = 120960, EH2H = 61452, EH2E = 120960;

  const float* x        = (const float*)d_in[0];
  const float* c_era    = (const float*)d_in[1];
  const float* c_h      = (const float*)d_in[2];
  const int*   e2h_src  = (const int*)d_in[3];
  const int*   e2h_dst  = e2h_src + EE2H;
  const float* e2h_attr = (const float*)d_in[4];
  const int*   h2h_src  = (const int*)d_in[5];
  const int*   h2h_dst  = h2h_src + EH2H;
  const float* h2h_attr = (const float*)d_in[6];
  const int*   h2e_src  = (const int*)d_in[7];
  const int*   h2e_dst  = h2e_src + EH2E;
  const float* h2e_attr = (const float*)d_in[8];
  const float* noise    = (const float*)d_in[9];
  const float* era_tr   = (const float*)d_in[10];
  const float* h_tr     = (const float*)d_in[11];
  const float* e2h_tr   = (const float*)d_in[12];
  const float* h2e_tr   = (const float*)d_in[13];
  const float* h2h_tr   = (const float*)d_in[14];

  // param leaf indices (depth-first, insertion order; each MLP = W0,b0,W1,b1[,g,be])
  enum { ENC_SRC=15, ENC_DST=21, ENC_EDGE=27, ENC_MSG=33, ENC_NODE=39,
         PROC_NOISE=45, PROC_EDGE=51, P0_MSG=57, P0_NODE=63, P1_MSG=69, P1_NODE=75,
         DEC_SRC=81, DEC_DST=87, DEC_EDGE=93, DEC_MSG=99, DEC_NODE=105, DEC_OUT=111 };

  char* wsb = (char*)d_ws; size_t off = 0;
  auto alloc = [&](size_t bytes) -> void* {
    void* p = wsb + off; off += (bytes + 255) & ~(size_t)255; return p;
  };
  // activation buffers padded to 128-row multiples (guard-free GEMM A loads)
  auto allocRows = [&](int rows, int cols) -> float* {
    size_t r = ((size_t)rows + 127) & ~(size_t)127;
    return (float*)alloc(r * (size_t)cols * 4);
  };

  auto conv_w = [&](int idx, int K, int N, int Kpad, int Npad) -> const unsigned short* {
    unsigned short* wt = (unsigned short*)alloc((size_t)Kpad * Npad * 2);
    int total = Kpad * Npad;
    k_convert_w<<<dim3((total + 255) / 256), dim3(256), 0, stream>>>(
        (const float*)d_in[idx], wt, K, N, Kpad, Npad);
    return wt;
  };

  auto gemm = [&](Seg a0, Seg a1, Seg a2, int nseg, const unsigned short* Wt,
                  const float* bias, float* outp, int M, int Nreal, int Npad,
                  int Kpad, int act) {
    dim3 blk(256);
    if (nseg == 3) {           // message MLPs: Kpad==768, Npad==256; LDS-staged B
      dim3 grid((M + 127) / 128, Npad / 64);
      k_gemm<3,1,768,4><<<grid, blk, 0, stream>>>(a0,a1,a2,Wt,bias,outp,M,Nreal,Kpad,Npad);
    } else if (nseg == 2) {    // layer-0 node/noise MLPs: always SiLU, Npad==256
      dim3 grid((M + 127) / 128, Npad / 64);
      k_gemm<2,1,0,4><<<grid, blk, 0, stream>>>(a0,a1,a2,Wt,bias,outp,M,Nreal,Kpad,Npad);
    } else if (Npad % 64 == 0) {
      dim3 grid((M + 127) / 128, Npad / 64);
      if (act == 1) k_gemm<1,1,0,4><<<grid, blk, 0, stream>>>(a0,a1,a2,Wt,bias,outp,M,Nreal,Kpad,Npad);
      else          k_gemm<1,0,0,4><<<grid, blk, 0, stream>>>(a0,a1,a2,Wt,bias,outp,M,Nreal,Kpad,Npad);
    } else {                   // final 96-wide output layer
      dim3 grid((M + 127) / 128, Npad / 32);
      if (act == 1) k_gemm<1,1,0,2><<<grid, blk, 0, stream>>>(a0,a1,a2,Wt,bias,outp,M,Nreal,Kpad,Npad);
      else          k_gemm<1,0,0,2><<<grid, blk, 0, stream>>>(a0,a1,a2,Wt,bias,outp,M,Nreal,Kpad,Npad);
    }
  };

  auto ln = [&](const float* xx, int gidx, const float* res, float* outp, int M) {
    k_layernorm<<<dim3((M + 7) / 8), dim3(256), 0, stream>>>(
        xx, (const float*)d_in[gidx], (const float*)d_in[gidx + 1], res, outp, M);
  };

  auto zero = [&](float* p, size_t n) {
    k_zero<<<dim3(2048), dim3(256), 0, stream>>>(p, n);
  };

  const Seg DSEG{nullptr, nullptr, 0, 0};

  // 2-layer MLP (K -> 256 silu -> 256) + LayerNorm (optional residual)
  auto run_mlp = [&](Seg a0, Seg a1, Seg a2, int nseg, int K, int Kpad, int pidx,
                     int M, float* tmp, float* raw, const float* res, float* lnout) {
    const unsigned short* W0 = conv_w(pidx, K, 256, Kpad, 256);
    gemm(a0, a1, a2, nseg, W0, (const float*)d_in[pidx + 1], tmp, M, 256, 256, Kpad, 1);
    const unsigned short* W1 = conv_w(pidx + 2, 256, 256, 256, 256);
    Seg t{tmp, nullptr, 256, 256};
    gemm(t, DSEG, DSEG, 1, W1, (const float*)d_in[pidx + 3], raw, M, 256, 256, 256, 0);
    ln(raw, pidx + 4, res, lnout, M);
  };

  // ---- workspace buffers (row-padded) ----
  float* xera     = allocRows(NERA, 224);
  float* xh       = allocRows(NH,   32);
  float* edgeIn   = allocRows(EE2H, 32);
  float* noisePad = allocRows(NH,   32);
  float* eBuf     = allocRows(EE2H, 256);
  float* tmpE     = allocRows(EE2H, 256);
  float* rawE     = allocRows(EE2H, 256);
  float* hs       = allocRows(NERA, 256);
  float* tmpN     = allocRows(NERA, 256);
  float* rawN     = allocRows(NERA, 256);
  float* agg      = allocRows(NERA, 256);
  float* hd       = allocRows(NH,   256);
  float* hdU      = allocRows(NH,   256);
  float* h        = allocRows(NH,   256);
  float* hA       = allocRows(NH,   256);
  float* hd2      = allocRows(NERA, 256);
  float* hd2b     = allocRows(NERA, 256);

  // h2h edge count is not a 128-multiple: padded gather-index copies
  const int EH2Hp = (EH2H + 127) & ~127;
  int* h2h_src_p = (int*)alloc((size_t)EH2Hp * 4);
  int* h2h_dst_p = (int*)alloc((size_t)EH2Hp * 4);
  k_pad_idx<<<dim3((EH2Hp + 255) / 256), dim3(256), 0, stream>>>(h2h_src, h2h_src_p, EH2H, EH2Hp);
  k_pad_idx<<<dim3((EH2Hp + 255) / 256), dim3(256), 0, stream>>>(h2h_dst, h2h_dst_p, EH2H, EH2Hp);

  // ---- feature construction ----
  k_build_xera<<<dim3((NERA * 224 + 255) / 256), dim3(256), 0, stream>>>(x, c_era, era_tr, xera, NERA);
  k_build_xh  <<<dim3((NH * 32 + 255) / 256),   dim3(256), 0, stream>>>(c_h, h_tr, xh, NH);

  // ================= ENCODER (era -> h) =================
  run_mlp(Seg{xera, nullptr, 224, 224}, DSEG, DSEG, 1, 212, 224, ENC_SRC, NERA, tmpN, hs, nullptr, hs);
  run_mlp(Seg{xh,   nullptr, 32,  32 }, DSEG, DSEG, 1, 12,  32,  ENC_DST, NH,   tmpN, hd, nullptr, hd);
  k_build_edgein<<<dim3((EE2H * 32 + 255) / 256), dim3(256), 0, stream>>>(e2h_attr, e2h_tr, edgeIn, EE2H);
  run_mlp(Seg{edgeIn, nullptr, 32, 32}, DSEG, DSEG, 1, 11, 32, ENC_EDGE, EE2H, tmpE, eBuf, nullptr, eBuf);

  // message block: m = LN(MLP([hs[src] | hd[dst] | e])); e += m; agg = segsum(m)
  run_mlp(Seg{hs, e2h_src, 256, 256}, Seg{hd, e2h_dst, 256, 256}, Seg{eBuf, nullptr, 256, 256},
          3, 768, 768, ENC_MSG, EE2H, tmpE, rawE, nullptr, rawE);
  zero(agg, (size_t)NH * 256);
  k_edge_scatter<<<dim3((EE2H + 7) / 8), dim3(256), 0, stream>>>(rawE, eBuf, agg, e2h_dst, EE2H);
  run_mlp(Seg{hd, nullptr, 256, 256}, Seg{agg, nullptr, 256, 256}, DSEG,
          2, 512, 512, ENC_NODE, NH, tmpN, rawN, hd, hdU);

  // ================= PROCESSOR (h -> h, 2 layers) =================
  k_pad_noise<<<dim3((NH * 32 + 255) / 256), dim3(256), 0, stream>>>(noise, noisePad, NH);
  run_mlp(Seg{hdU, nullptr, 256, 256}, Seg{noisePad, nullptr, 32, 32}, DSEG,
          2, 260, 288, PROC_NOISE, NH, tmpN, h, nullptr, h);
  k_build_edgein<<<dim3((EH2H * 32 + 255) / 256), dim3(256), 0, stream>>>(h2h_attr, h2h_tr, edgeIn, EH2H);
  run_mlp(Seg{edgeIn, nullptr, 32, 32}, DSEG, DSEG, 1, 11, 32, PROC_EDGE, EH2H, tmpE, eBuf, nullptr, eBuf);

  float* cur = h; float* nxt = hA;
  const int MSGI[2] = {P0_MSG, P1_MSG};
  const int NODI[2] = {P0_NODE, P1_NODE};
  for (int L = 0; L < 2; ++L) {
    run_mlp(Seg{cur, h2h_src_p, 256, 256}, Seg{cur, h2h_dst_p, 256, 256}, Seg{eBuf, nullptr, 256, 256},
            3, 768, 768, MSGI[L], EH2H, tmpE, rawE, nullptr, rawE);
    zero(agg, (size_t)NH * 256);
    k_edge_scatter<<<dim3((EH2H + 7) / 8), dim3(256), 0, stream>>>(rawE, eBuf, agg, h2h_dst, EH2H);
    run_mlp(Seg{cur, nullptr, 256, 256}, Seg{agg, nullptr, 256, 256}, DSEG,
            2, 512, 512, NODI[L], NH, tmpN, rawN, cur, nxt);
    float* t = cur; cur = nxt; nxt = t;
  }
  // cur = final processor state, nxt is free

  // ================= DECODER (h -> era) =================
  float* hs2 = nxt;
  run_mlp(Seg{cur, nullptr, 256, 256}, DSEG, DSEG, 1, 256, 256, DEC_SRC, NH,   tmpN, hs2, nullptr, hs2);
  run_mlp(Seg{hs,  nullptr, 256, 256}, DSEG, DSEG, 1, 256, 256, DEC_DST, NERA, tmpN, hd2, nullptr, hd2);
  k_build_edgein<<<dim3((EH2E * 32 + 255) / 256), dim3(256), 0, stream>>>(h2e_attr, h2e_tr, edgeIn, EH2E);
  run_mlp(Seg{edgeIn, nullptr, 32, 32}, DSEG, DSEG, 1, 11, 32, DEC_EDGE, EH2E, tmpE, eBuf, nullptr, eBuf);

  run_mlp(Seg{hs2, h2e_src, 256, 256}, Seg{hd2, h2e_dst, 256, 256}, Seg{eBuf, nullptr, 256, 256},
          3, 768, 768, DEC_MSG, EH2E, tmpE, rawE, nullptr, rawE);
  zero(agg, (size_t)NERA * 256);
  k_edge_scatter<<<dim3((EH2E + 7) / 8), dim3(256), 0, stream>>>(rawE, eBuf, agg, h2e_dst, EH2E);
  run_mlp(Seg{hd2, nullptr, 256, 256}, Seg{agg, nullptr, 256, 256}, DSEG,
          2, 512, 512, DEC_NODE, NERA, tmpN, rawN, hd2, hd2b);

  // output MLP (no LN): 256 -> 256 (silu) -> 88 (padded to 96)
  const unsigned short* Wo0 = conv_w(DEC_OUT, 256, 256, 256, 256);
  gemm(Seg{hd2b, nullptr, 256, 256}, DSEG, DSEG, 1, Wo0, (const float*)d_in[DEC_OUT + 1],
       tmpN, NERA, 256, 256, 256, 1);
  const unsigned short* Wo1 = conv_w(DEC_OUT + 2, 256, 88, 256, 96);
  float* out96 = rawN;   // reuse, ld = 96
  gemm(Seg{tmpN, nullptr, 256, 256}, DSEG, DSEG, 1, Wo1, (const float*)d_in[DEC_OUT + 3],
       out96, NERA, 88, 96, 256, 0);
  k_copy_out<<<dim3((NERA * 88 + 255) / 256), dim3(256), 0, stream>>>(out96, (float*)d_out, NERA);
}
